// BoxTransformerEncoderLayer_420906795022
// MI455X (gfx1250) — matrix operand order repaired and checked by reading, not compiled, not run on hardware
//
#include <hip/hip_runtime.h>
#include <cstdint>
#include <cstddef>

// ---------------- problem constants ----------------
#define B_      2
#define S_      11253
#define D_      256
#define NHEAD_  8
#define NLEVEL_ 4
#define NPOINT_ 4
#define HD_     32
#define DFF_    1024

typedef __bf16 bf16_t;
typedef __attribute__((ext_vector_type(16))) __bf16 v16bf;
typedef __attribute__((ext_vector_type(8)))  __bf16 v8bf;
typedef __attribute__((ext_vector_type(8)))  float  v8f;

static __device__ __forceinline__ bf16_t f2bf(float f) {
  union { float f; unsigned u; } in; in.f = f;
  unsigned u = (in.u + 0x7FFFu + ((in.u >> 16) & 1u)) >> 16;   // RNE
  union { unsigned short s; bf16_t b; } out; out.s = (unsigned short)u;
  return out.b;
}

// ---- CDNA5 async global->LDS (ASYNCcnt-tracked), 16B per lane ----
static __device__ __forceinline__ void async_copy16(unsigned lds_off, const void* gptr) {
  asm volatile("global_load_async_to_lds_b128 %0, %1, off"
               :: "v"(lds_off), "v"((unsigned long long)(size_t)gptr)
               : "memory");
}
static __device__ __forceinline__ void wait_async0() {
#if __has_builtin(__builtin_amdgcn_s_wait_asynccnt)
  __builtin_amdgcn_s_wait_asynccnt(0);
#else
  asm volatile("s_wait_asynccnt 0x0" ::: "memory");
#endif
}

// ---------------- generic f32 -> bf16 convert ----------------
__global__ __launch_bounds__(256) void cvt_k(const float* __restrict__ src,
                                             bf16_t* __restrict__ dst, int n) {
  int i = blockIdx.x * 256 + threadIdx.x;
  if (i < n) dst[i] = f2bf(src[i]);
}

// q = src + pos (bf16), src copy (bf16)
__global__ __launch_bounds__(256) void prep_k(const float* __restrict__ src,
                                              const float* __restrict__ pos,
                                              bf16_t* __restrict__ src_bf,
                                              bf16_t* __restrict__ q_bf,
                                              long long n) {
  long long i = (long long)blockIdx.x * 256 + threadIdx.x;
  if (i < n) {
    float s = src[i];
    src_bf[i] = f2bf(s);
    q_bf[i]   = f2bf(s + pos[i]);
  }
}

// ---------------- WMMA bf16 GEMM:  C[M,N] = A[M,K] * W[N,K]^T (+bias)(+res)(relu?) -------
// Block tile 128(M) x 128(N), K-step 32, async double-buffered LDS.
// 8 waves: wm = wid>>1 (4 x 32 rows), wn = wid&1 (2 x 64 cols).
// Each wave: 2 A-frags x 4 B-frags = 8 WMMAs per K-step.
#define LSTR 40   // LDS row stride in elements (80B, 16B aligned)
template<bool RELU, bool BF16OUT>
__global__ __launch_bounds__(256)
void gemm_bf16(const bf16_t* __restrict__ A, const bf16_t* __restrict__ W,
               const float* __restrict__ bias, const float* __restrict__ residual,
               void* __restrict__ Cout, int M, int N, int K)
{
  __shared__ bf16_t Alds[2][128 * LSTR];
  __shared__ bf16_t Blds[2][128 * LSTR];

  const int tid  = threadIdx.x;
  const int lane = tid & 31;
  const int wid  = tid >> 5;
  const int wm   = wid >> 1;        // 0..3
  const int wn   = wid & 1;         // 0..1
  const int half = lane >> 4;       // ISA 16-bit fragment half select
  const int l15  = lane & 15;

  const int m_blk = blockIdx.y * 128;
  const int n_blk = blockIdx.x * 128;

  v8f acc[2][4] = {};

  // staging: 256 threads, 2 threads per row, 16-element (32B) chunk each
  const int row    = tid >> 1;           // 0..127
  const int cchunk = (tid & 1) * 16;     // 0 or 16 elements

  const int ga_row = min(m_blk + row, M - 1);   // clamp: pad rows duplicate last row
  const bf16_t* aptr = A + (size_t)ga_row * K + cchunk;
  const bf16_t* bptr = W + (size_t)(n_blk + row) * K + cchunk;

  const unsigned a_lds = (unsigned)(size_t)&Alds[0][row * LSTR + cchunk];
  const unsigned b_lds = (unsigned)(size_t)&Blds[0][row * LSTR + cchunk];
  const unsigned bufstride = (unsigned)(128 * LSTR * sizeof(bf16_t));

  // prologue: async-stage tile 0
  async_copy16(a_lds,      aptr);
  async_copy16(a_lds + 16, aptr + 8);
  async_copy16(b_lds,      bptr);
  async_copy16(b_lds + 16, bptr + 8);

  const int nk = K >> 5;
  for (int k = 0; k < nk; ++k) {
    wait_async0();          // this wave's portion of tile k is in LDS
    __syncthreads();        // whole tile k resident; previous compute done

    if (k + 1 < nk) {       // async-stage tile k+1 into the other buffer
      const bf16_t* ap = aptr + (size_t)(k + 1) * 32;
      const bf16_t* bp = bptr + (size_t)(k + 1) * 32;
      const unsigned ao = a_lds + ((k + 1) & 1) * bufstride;
      const unsigned bo = b_lds + ((k + 1) & 1) * bufstride;
      async_copy16(ao,      ap);
      async_copy16(ao + 16, ap + 8);
      async_copy16(bo,      bp);
      async_copy16(bo + 16, bp + 8);
      if (k + 2 < nk) {     // L2 hint two tiles ahead
        __builtin_prefetch(ap + 32, 0, 2);
        __builtin_prefetch(bp + 32, 0, 2);
      }
    }

    const bf16_t* Ab = Alds[k & 1];
    const bf16_t* Bb = Blds[k & 1];

    // A fragments: lane holds row (wm*32 + ms*16 + l15); K = {half*8..+7, 16+half*8..+7}
    v16bf afrag[2];
    #pragma unroll
    for (int ms = 0; ms < 2; ++ms) {
      const bf16_t* ap2 = &Ab[(wm * 32 + ms * 16 + l15) * LSTR + half * 8];
      v8bf lo = *(const v8bf*)(ap2);
      v8bf hi = *(const v8bf*)(ap2 + 16);
      #pragma unroll
      for (int i = 0; i < 8; ++i) { afrag[ms][i] = lo[i]; afrag[ms][i + 8] = hi[i]; }
    }

    #pragma unroll
    for (int ns = 0; ns < 4; ++ns) {
      // B fragment: lane holds out-col; K = {half*8..+7, 16+half*8..+7}
      const bf16_t* bp2 = &Bb[(wn * 64 + ns * 16 + l15) * LSTR + half * 8];
      v8bf lo = *(const v8bf*)(bp2);
      v8bf hi = *(const v8bf*)(bp2 + 16);
      v16bf bfrag;
      #pragma unroll
      for (int i = 0; i < 8; ++i) { bfrag[i] = lo[i]; bfrag[i + 8] = hi[i]; }
      acc[0][ns] = __builtin_amdgcn_wmma_f32_16x16x32_bf16(
          false, afrag[0], false, bfrag, (short)0, acc[0][ns], false, false);
      acc[1][ns] = __builtin_amdgcn_wmma_f32_16x16x32_bf16(
          false, afrag[1], false, bfrag, (short)0, acc[1][ns], false, false);
    }
    // no trailing barrier needed: next iteration's wait+barrier orders buffer reuse
  }

  // epilogue: D element r -> M = r + half*8, N = l15
  #pragma unroll
  for (int ms = 0; ms < 2; ++ms) {
    #pragma unroll
    for (int ns = 0; ns < 4; ++ns) {
      const int n = n_blk + wn * 64 + ns * 16 + l15;
      const float bv = bias ? bias[n] : 0.0f;
      #pragma unroll
      for (int r = 0; r < 8; ++r) {
        const int m = m_blk + wm * 32 + ms * 16 + half * 8 + r;
        if (m < M) {
          float v = acc[ms][ns][r] + bv;
          if (residual) v += residual[(size_t)m * N + n];
          if (RELU) v = fmaxf(v, 0.0f);
          if (BF16OUT) ((bf16_t*)Cout)[(size_t)m * N + n] = f2bf(v);
          else         ((float*)Cout)[(size_t)m * N + n]  = v;
        }
      }
    }
  }
}

// ---------------- box-attention sampling: one wave per (b,s,head), lane = channel -------
static __device__ __forceinline__ float fetchv(const float* __restrict__ value,
                                               int b, int s0, int h, int lane,
                                               int ix, int iy, int W, int H) {
  bool valid = (ix >= 0) & (ix < W) & (iy >= 0) & (iy < H);
  int cx = min(max(ix, 0), W - 1);
  int cy = min(max(iy, 0), H - 1);
  size_t idx = (size_t)s0 + (size_t)cy * W + cx;
  float g = value[(((size_t)b * S_ + idx) * NHEAD_ + h) * HD_ + lane];
  return valid ? g : 0.0f;
}

__global__ __launch_bounds__(256)
void sample_k(const float* __restrict__ off, const float* __restrict__ alog,
              const float* __restrict__ refw, const float* __restrict__ vr,
              const float* __restrict__ value, bf16_t* __restrict__ samp)
{
  const int   LVL_HW[4] = {92, 46, 23, 12};
  const int   LVL_S0[4] = {0, 8464, 10580, 11109};
  const float KX[4] = {-0.25f, 0.25f, -0.25f, 0.25f};
  const float KY[4] = {-0.25f, -0.25f, 0.25f, 0.25f};

  const int gid  = blockIdx.x * 8 + (threadIdx.x >> 5);
  const int lane = threadIdx.x & 31;
  if (gid >= B_ * S_ * NHEAD_) return;

  const int h  = gid & (NHEAD_ - 1);
  const int bs = gid >> 3;
  const int s  = bs % S_;
  const int b  = bs / S_;
  const size_t m = (size_t)b * S_ + s;

  const float4 r = ((const float4*)refw)[m];

  // softmax over 16 logits held in lanes 0..15
  float logit = (lane < 16) ? alog[m * 128 + h * 16 + lane] : -3.0e38f;
  float amax = logit;
  #pragma unroll
  for (int x = 8; x >= 1; x >>= 1) amax = fmaxf(amax, __shfl_xor(amax, x, 32));
  float ev = __expf(logit - amax);
  float esum = ev;
  #pragma unroll
  for (int x = 8; x >= 1; x >>= 1) esum += __shfl_xor(esum, x, 32);
  const float wgt = ev / esum;

  float accv = 0.0f;
  #pragma unroll
  for (int lvl = 0; lvl < NLEVEL_; ++lvl) {
    const int Hh = LVL_HW[lvl], Ww = LVL_HW[lvl], s0 = LVL_S0[lvl];
    const float vrx = vr[(b * NLEVEL_ + lvl) * 2 + 0];
    const float vry = vr[(b * NLEVEL_ + lvl) * 2 + 1];
    const float* ob = off + m * 128 + (h * NLEVEL_ + lvl) * 4;
    const float cx = r.x + ob[0] * 0.125f * r.z;
    const float cy = r.y + ob[1] * 0.125f * r.w;
    const float bw = fmaxf(r.z + ob[2] * 0.125f * r.z, 0.0f);
    const float bh = fmaxf(r.w + ob[3] * 0.125f * r.w, 0.0f);
    #pragma unroll
    for (int pt = 0; pt < NPOINT_; ++pt) {
      const float gx = (cx + KX[pt] * bw) * vrx;
      const float gy = (cy + KY[pt] * bh) * vry;
      const float px = gx * Ww - 0.5f;
      const float py = gy * Hh - 0.5f;
      const float xf = floorf(px), yf = floorf(py);
      const float wx = px - xf,   wy = py - yf;
      const int ix = (int)xf, iy = (int)yf;
      const float g00 = fetchv(value, b, s0, h, lane, ix,     iy,     Ww, Hh);
      const float g01 = fetchv(value, b, s0, h, lane, ix + 1, iy,     Ww, Hh);
      const float g10 = fetchv(value, b, s0, h, lane, ix,     iy + 1, Ww, Hh);
      const float g11 = fetchv(value, b, s0, h, lane, ix + 1, iy + 1, Ww, Hh);
      const float bil = (g00 * (1.0f - wx) + g01 * wx) * (1.0f - wy)
                      + (g10 * (1.0f - wx) + g11 * wx) * wy;
      accv += bil * __shfl(wgt, lvl * NPOINT_ + pt, 32);
    }
  }
  samp[m * D_ + h * HD_ + lane] = f2bf(accv);
}

// ---------------- layernorm: one wave per row of 256 --------------------------------
__global__ __launch_bounds__(256)
void layernorm_k(const float* __restrict__ in, const float* __restrict__ gw,
                 const float* __restrict__ gb, float* __restrict__ outf,
                 bf16_t* __restrict__ outbf, int M)
{
  const int row  = blockIdx.x * 8 + (threadIdx.x >> 5);
  const int lane = threadIdx.x & 31;
  if (row >= M) return;
  const float* p = in + (size_t)row * D_ + lane * 8;
  float e[8];
  float4 a0 = *(const float4*)p;
  float4 a1 = *(const float4*)(p + 4);
  e[0]=a0.x; e[1]=a0.y; e[2]=a0.z; e[3]=a0.w;
  e[4]=a1.x; e[5]=a1.y; e[6]=a1.z; e[7]=a1.w;
  float s = 0.0f;
  #pragma unroll
  for (int i = 0; i < 8; ++i) s += e[i];
  #pragma unroll
  for (int x = 16; x >= 1; x >>= 1) s += __shfl_xor(s, x, 32);
  const float mu = s * (1.0f / D_);
  float vs = 0.0f;
  #pragma unroll
  for (int i = 0; i < 8; ++i) { float d = e[i] - mu; vs += d * d; }
  #pragma unroll
  for (int x = 16; x >= 1; x >>= 1) vs += __shfl_xor(vs, x, 32);
  const float rstd = rsqrtf(vs * (1.0f / D_) + 1e-5f);
  #pragma unroll
  for (int i = 0; i < 8; ++i) {
    const int col = lane * 8 + i;
    const float o = (e[i] - mu) * rstd * gw[col] + gb[col];
    if (outf)  outf[(size_t)row * D_ + col]  = o;
    if (outbf) outbf[(size_t)row * D_ + col] = f2bf(o);
  }
}

// ---------------- host orchestration ----------------
extern "C" void kernel_launch(void* const* d_in, const int* in_sizes, int n_in,
                              void* d_out, int out_size, void* d_ws, size_t ws_size,
                              hipStream_t stream)
{
  (void)in_sizes; (void)n_in; (void)out_size; (void)ws_size;
  const float* src     = (const float*)d_in[0];
  const float* pos     = (const float*)d_in[1];
  const float* vr      = (const float*)d_in[5];
  const float* refw    = (const float*)d_in[6];
  const float* value_w = (const float*)d_in[7];
  const float* value_b = (const float*)d_in[8];
  const float* out_w   = (const float*)d_in[9];
  const float* out_b   = (const float*)d_in[10];
  const float* box_w   = (const float*)d_in[11];
  const float* box_b   = (const float*)d_in[12];
  const float* attn_w  = (const float*)d_in[13];
  const float* attn_b  = (const float*)d_in[14];
  const float* lin1_w  = (const float*)d_in[15];
  const float* lin1_b  = (const float*)d_in[16];
  const float* lin2_w  = (const float*)d_in[17];
  const float* lin2_b  = (const float*)d_in[18];
  const float* n1w = (const float*)d_in[19];
  const float* n1b = (const float*)d_in[20];
  const float* n2w = (const float*)d_in[21];
  const float* n2b = (const float*)d_in[22];

  const int M = B_ * S_;                 // 22506 rows

  char* ws = (char*)d_ws;
  size_t cur = 0;
  auto alloc = [&](size_t bytes) -> void* {
    void* p = ws + cur; cur += (bytes + 255) & ~(size_t)255; return p;
  };

  bf16_t* wv   = (bf16_t*)alloc((size_t)D_ * D_ * 2);
  bf16_t* wbox = (bf16_t*)alloc((size_t)128 * D_ * 2);
  bf16_t* watt = (bf16_t*)alloc((size_t)128 * D_ * 2);
  bf16_t* wout = (bf16_t*)alloc((size_t)D_ * D_ * 2);
  bf16_t* w1   = (bf16_t*)alloc((size_t)DFF_ * D_ * 2);
  bf16_t* w2   = (bf16_t*)alloc((size_t)D_ * DFF_ * 2);
  bf16_t* src_bf  = (bf16_t*)alloc((size_t)M * D_ * 2);
  bf16_t* q_bf    = (bf16_t*)alloc((size_t)M * D_ * 2);
  float*  value_f = (float*) alloc((size_t)M * D_ * 4);
  float*  off_f   = (float*) alloc((size_t)M * 128 * 4);
  float*  attn_f  = (float*) alloc((size_t)M * 128 * 4);
  bf16_t* samp_bf = (bf16_t*)alloc((size_t)M * D_ * 2);
  float*  res1_f  = (float*) alloc((size_t)M * D_ * 4);   // reused for y=x+ff
  float*  x_f     = (float*) alloc((size_t)M * D_ * 4);
  bf16_t* x_bf    = (bf16_t*)alloc((size_t)M * D_ * 2);
  bf16_t* h_bf    = (bf16_t*)alloc((size_t)M * DFF_ * 2);

  auto cvt = [&](const float* s, bf16_t* d, int n) {
    cvt_k<<<(n + 255) / 256, 256, 0, stream>>>(s, d, n);
  };
  cvt(value_w, wv,   D_ * D_);
  cvt(box_w,   wbox, 128 * D_);
  cvt(attn_w,  watt, 128 * D_);
  cvt(out_w,   wout, D_ * D_);
  cvt(lin1_w,  w1,   DFF_ * D_);
  cvt(lin2_w,  w2,   D_ * DFF_);

  const long long nElems = (long long)M * D_;
  prep_k<<<(int)((nElems + 255) / 256), 256, 0, stream>>>(src, pos, src_bf, q_bf, nElems);

  const int mb = (M + 127) / 128;
  // value = src @ value_w^T + b
  gemm_bf16<false,false><<<dim3(D_ / 128, mb), 256, 0, stream>>>(
      src_bf, wv, value_b, nullptr, value_f, M, D_, D_);
  // off = q @ box_w^T + b ; attn_logits = q @ attn_w^T + b
  gemm_bf16<false,false><<<dim3(1, mb), 256, 0, stream>>>(
      q_bf, wbox, box_b, nullptr, off_f, M, 128, D_);
  gemm_bf16<false,false><<<dim3(1, mb), 256, 0, stream>>>(
      q_bf, watt, attn_b, nullptr, attn_f, M, 128, D_);

  // bilinear box sampling, softmax-weighted
  sample_k<<<(B_ * S_ * NHEAD_ + 7) / 8, 256, 0, stream>>>(
      off_f, attn_f, refw, vr, value_f, samp_bf);

  // out projection + residual(src)
  gemm_bf16<false,false><<<dim3(D_ / 128, mb), 256, 0, stream>>>(
      samp_bf, wout, out_b, src, res1_f, M, D_, D_);
  // norm1 -> x (f32 + bf16)
  layernorm_k<<<(M + 7) / 8, 256, 0, stream>>>(res1_f, n1w, n1b, x_f, x_bf, M);

  // ffn
  gemm_bf16<true,true><<<dim3(DFF_ / 128, mb), 256, 0, stream>>>(
      x_bf, w1, lin1_b, nullptr, h_bf, M, DFF_, D_);
  gemm_bf16<false,false><<<dim3(D_ / 128, mb), 256, 0, stream>>>(
      h_bf, w2, lin2_b, x_f, res1_f, M, D_, DFF_);
  // norm2 -> d_out
  layernorm_k<<<(M + 7) / 8, 256, 0, stream>>>(res1_f, n2w, n2b, (float*)d_out, nullptr, M);
}
